// Hierarchical_Generator_9887014715850
// MI455X (gfx1250) — matrix operand order repaired and checked by reading, not compiled, not run on hardware
//
#include <hip/hip_runtime.h>

// ---------------- problem constants (from reference) ----------------
#define N_ROWS 10000
#define MPAD   10016          // rows padded to multiple of 32 (wave row-pair tile)
#define KPAD   10016          // K padded to multiple of 32 (WMMA K)
#define FT     128
#define DT     64
#define GHID   128
#define GC     32
#define NINF   95             // GC + (FT - DT - 1)
#define MH1    64
#define MH2    32
#define NNK    100

// ---------------- WMMA types ----------------
typedef __bf16 bf16_t;
typedef bf16_t v16bf __attribute__((ext_vector_type(16)));
typedef float  v8f   __attribute__((ext_vector_type(8)));

struct __attribute__((aligned(16))) U128 { unsigned int x, y, z, w; };
union Frag { U128 u[2]; v16bf v; };

__device__ __forceinline__ unsigned short f32_to_bf16(float f) {
  unsigned int u = __float_as_uint(f);
  u = u + 0x7FFFu + ((u >> 16) & 1u);     // round-to-nearest-even
  return (unsigned short)(u >> 16);
}

// ---------------- 1) adj fp32 -> bf16, padded [MPAD][KPAD], zeros in pads ---
__global__ void conv_adj_bf16(const float* __restrict__ adj,
                              unsigned short* __restrict__ adjB) {
  const size_t total = (size_t)MPAD * KPAD;
  for (size_t idx = (size_t)blockIdx.x * blockDim.x + threadIdx.x;
       idx < total; idx += (size_t)gridDim.x * blockDim.x) {
    size_t r = idx / KPAD;
    int    c = (int)(idx - r * KPAD);
    float  v = (r < N_ROWS && c < N_ROWS) ? adj[r * (size_t)N_ROWS + c] : 0.0f;
    adjB[idx] = f32_to_bf16(v);
  }
}

// ---------------- 2) B = A[0:K, 0:Kin] @ W  written in WMMA B-fragment layout
// layout: frag[(k>>5) * Nc * 32 + n * 32 + (k&31)], rows k >= K are zero.
__global__ void make_b_frag(const float* __restrict__ A, int lda, int K, int Kin,
                            const float* __restrict__ W, int Nc,
                            unsigned short* __restrict__ out) {
  int idx = blockIdx.x * blockDim.x + threadIdx.x;
  int total = KPAD * Nc;
  if (idx >= total) return;
  int k = idx / Nc;
  int n = idx - k * Nc;
  float acc = 0.0f;
  if (k < K) {
    const float* arow = A + (size_t)k * lda;
    for (int i = 0; i < Kin; ++i) acc += arow[i] * W[i * Nc + n];
  }
  out[((size_t)(k >> 5) * Nc + n) * 32 + (k & 31)] = f32_to_bf16(acc);
}

// ---------------- 3) H = relu(adjB @ Bfrag + bias), bf16 WMMA, f32 accum ----
// 8 waves/block arranged as (8/CG row-pair groups) x (CG column groups).
// Each wave: 2 row-tiles (32 rows) x NCT col-tiles (NCT*16 cols).
// Live VGPRs ~ 64 acc + 16 A + <=8*NCT B: no spills by construction.
// Per ISA 16-bit operand layouts:
//   A lane<16: row = r0+lane,     K {0..7} then {16..23}
//   A lane>=16: row = r0+lane-16, K {8..15} then {24..31}
//   B lane<16: col = n0+lane, K {0..7} then {16..23};  lane>=16: K {8..15} then {24..31}
template <int NC, int NCT>
__global__ void __launch_bounds__(256)
gemm_wmma_relu(const unsigned short* __restrict__ adjB,
               const unsigned short* __restrict__ Bfrag,
               const float* __restrict__ bias,
               float* __restrict__ H, int M) {
  constexpr int CG = NC / (16 * NCT);   // column groups per block
  constexpr int RG = 8 / CG;            // row-pair groups per block
  const int wave = threadIdx.x >> 5;
  const int lane = threadIdx.x & 31;
  const int rp   = blockIdx.x * RG + (wave / CG);   // 32-row tile index
  const int cg   = wave % CG;
  if (rp >= (MPAD >> 5)) return;        // wave-uniform: EXEC stays all-ones
  const int r0   = rp << 5;
  const int half = lane >> 4;
  const int lrow = lane & 15;

  const unsigned short* Arow0 = adjB + (size_t)(r0 + lrow) * KPAD + half * 8;
  const unsigned short* Arow1 = Arow0 + (size_t)16 * KPAD;
  const int    c0    = cg * (NCT * 16);
  const size_t bbase = ((size_t)c0 + lrow) * 32 + half * 8;

  v8f acc[2][NCT];
#pragma unroll
  for (int rt = 0; rt < 2; ++rt)
#pragma unroll
    for (int t = 0; t < NCT; ++t)
      acc[rt][t] = (v8f){0.f,0.f,0.f,0.f,0.f,0.f,0.f,0.f};

  const int ksteps = KPAD >> 5;         // 313
#pragma unroll 1
  for (int kc = 0; kc < ksteps; ++kc) {
    if (kc + 4 < ksteps) {              // stream A ahead (global_prefetch_b8)
      __builtin_prefetch((const void*)(Arow0 + (size_t)(kc + 4) * 32), 0, 1);
      __builtin_prefetch((const void*)(Arow1 + (size_t)(kc + 4) * 32), 0, 1);
    }
    Frag a0, a1;
    a0.u[0] = *(const U128*)(Arow0 + (size_t)kc * 32);
    a0.u[1] = *(const U128*)(Arow0 + (size_t)kc * 32 + 16);
    a1.u[0] = *(const U128*)(Arow1 + (size_t)kc * 32);
    a1.u[1] = *(const U128*)(Arow1 + (size_t)kc * 32 + 16);

    const unsigned short* Bk = Bfrag + (size_t)kc * NC * 32 + bbase;
#pragma unroll
    for (int t = 0; t < NCT; ++t) {
      Frag b;
      b.u[0] = *(const U128*)(Bk + (size_t)t * 512);
      b.u[1] = *(const U128*)(Bk + (size_t)t * 512 + 16);
      acc[0][t] = __builtin_amdgcn_wmma_f32_16x16x32_bf16(
          false, a0.v, false, b.v, (short)0, acc[0][t], false, false);
      acc[1][t] = __builtin_amdgcn_wmma_f32_16x16x32_bf16(
          false, a1.v, false, b.v, (short)0, acc[1][t], false, false);
    }
  }

  // epilogue: lane<16 -> (row r0+v, col), lane>=16 -> (row r0+8+v, col)
#pragma unroll
  for (int rt = 0; rt < 2; ++rt) {
#pragma unroll
    for (int t = 0; t < NCT; ++t) {
      int   col = c0 + t * 16 + lrow;
      float bv  = bias[col];
#pragma unroll
      for (int v = 0; v < 8; ++v) {
        int row = r0 + rt * 16 + half * 8 + v;
        if (row < M) {
          float val = acc[rt][t][v] + bv;
          H[(size_t)row * NC + col] = val > 0.0f ? val : 0.0f;
        }
      }
    }
  }
}

// ---------------- 4) MLP head ----------------------------------------------
__global__ void mlp_head(const float* __restrict__ h3, const float* __restrict__ x,
                         const float* __restrict__ mw1, const float* __restrict__ mb1,
                         const float* __restrict__ mw2, const float* __restrict__ mb2,
                         const float* __restrict__ mw3, const float* __restrict__ mb3,
                         float* __restrict__ s, int n) {
  __shared__ float W1[NINF * MH1];
  __shared__ float B1[MH1];
  __shared__ float W2[MH1 * MH2];
  __shared__ float B2[MH2];
  __shared__ float W3[MH2];
  __shared__ float B3;
  int tid = threadIdx.x;
  for (int i = tid; i < NINF * MH1; i += blockDim.x) W1[i] = mw1[i];
  for (int i = tid; i < MH1;        i += blockDim.x) B1[i] = mb1[i];
  for (int i = tid; i < MH1 * MH2;  i += blockDim.x) W2[i] = mw2[i];
  for (int i = tid; i < MH2;        i += blockDim.x) B2[i] = mb2[i];
  for (int i = tid; i < MH2;        i += blockDim.x) W3[i] = mw3[i];
  if (tid == 0) B3 = mb3[0];
  __syncthreads();

  int row = blockIdx.x * blockDim.x + tid;
  if (row >= n) return;

  float feat[NINF];
  for (int j = 0; j < GC; ++j)          feat[j]      = h3[(size_t)row * GC + j];
  for (int j = 0; j < FT - DT - 1; ++j) feat[GC + j] = x[(size_t)row * FT + DT + j];

  float m1v[MH1];
  for (int o = 0; o < MH1; ++o) {
    float a = B1[o];
    for (int i = 0; i < NINF; ++i) a += feat[i] * W1[i * MH1 + o];
    m1v[o] = a > 0.0f ? a : 0.0f;
  }
  float m2v[MH2];
  for (int o = 0; o < MH2; ++o) {
    float a = B2[o];
    for (int i = 0; i < MH1; ++i) a += m1v[i] * W2[i * MH2 + o];
    m2v[o] = a > 0.0f ? a : 0.0f;
  }
  float a = B3;
  for (int i = 0; i < MH2; ++i) a += m2v[i] * W3[i];
  s[row] = a;
}

// ---------------- 5) reductions + top-k threshold + mask -------------------
__global__ void reduce_min_k(const float* __restrict__ v, int n,
                             float* __restrict__ outp) {
  __shared__ float sm[256];
  int tid = threadIdx.x;
  float m = 3.402823466e38f;
  for (int i = blockIdx.x * blockDim.x + tid; i < n; i += gridDim.x * blockDim.x)
    m = fminf(m, v[i]);
  sm[tid] = m;
  __syncthreads();
  for (int st = 128; st > 0; st >>= 1) {
    if (tid < st) sm[tid] = fminf(sm[tid], sm[tid + st]);
    __syncthreads();
  }
  if (tid == 0) outp[blockIdx.x] = sm[0];
}

__global__ void sprime_k(const float* __restrict__ s, const float* __restrict__ x,
                         const float* __restrict__ minv, int n,
                         float* __restrict__ sp) {
  int i = blockIdx.x * blockDim.x + threadIdx.x;
  if (i >= n) return;
  float g = x[(size_t)i * FT + (FT - 1)];
  sp[i] = (g == 0.0f) ? *minv : s[i];
}

// threshold = value with exactly NNK elements ranked above it (ties by index),
// i.e. top_k(s, NNK+1)[NNK]
__global__ void rank_thr_k(const float* __restrict__ sp, int n,
                           float* __restrict__ thr) {
  int i = blockIdx.x * blockDim.x + threadIdx.x;
  if (i >= n) return;
  float si = sp[i];
  int c = 0;
  for (int j = 0; j < n; ++j) {
    float sj = sp[j];
    c += (sj > si) || (sj == si && j < i);
  }
  if (c == NNK) *thr = si;
}

__global__ void final_out_k(const float* __restrict__ sp,
                            const float* __restrict__ thr, int n,
                            float* __restrict__ out) {
  int i = blockIdx.x * blockDim.x + threadIdx.x;
  if (i >= n) return;
  float t = *thr;
  float v = sp[i];
  out[i] = (v > t) ? v * (1.0f / v) : 0.0f;
}

// ---------------- host: orchestrate on `stream` ----------------------------
extern "C" void kernel_launch(void* const* d_in, const int* in_sizes, int n_in,
                              void* d_out, int out_size, void* d_ws, size_t ws_size,
                              hipStream_t stream) {
  (void)in_sizes; (void)n_in; (void)out_size; (void)ws_size;
  const float* x   = (const float*)d_in[0];
  const float* adj = (const float*)d_in[1];
  const float* gw1 = (const float*)d_in[2];
  const float* gb1 = (const float*)d_in[3];
  const float* gw2 = (const float*)d_in[4];
  const float* gb2 = (const float*)d_in[5];
  const float* gw3 = (const float*)d_in[6];
  const float* gb3 = (const float*)d_in[7];
  const float* mw1 = (const float*)d_in[8];
  const float* mb1 = (const float*)d_in[9];
  const float* mw2 = (const float*)d_in[10];
  const float* mb2 = (const float*)d_in[11];
  const float* mw3 = (const float*)d_in[12];
  const float* mb3 = (const float*)d_in[13];
  float* out = (float*)d_out;

  char* ws = (char*)d_ws;
  size_t off = 0;
  unsigned short* adjB  = (unsigned short*)(ws + off); off += (size_t)MPAD * KPAD * 2;
  unsigned short* zfrag = (unsigned short*)(ws + off); off += (size_t)KPAD * 128 * 2;
  float* h       = (float*)(ws + off); off += (size_t)N_ROWS * 128 * 4;
  float* h3      = (float*)(ws + off); off += (size_t)N_ROWS * GC * 4;
  float* s       = (float*)(ws + off); off += (size_t)N_ROWS * 4;
  float* sp      = (float*)(ws + off); off += (size_t)N_ROWS * 4;
  float* partial = (float*)(ws + off); off += 64 * 4;
  float* minv    = (float*)(ws + off); off += 16;
  float* thr     = (float*)(ws + off); off += 16;

  const int rowPairs = MPAD / 32;                      // 313
  const int grid128  = (rowPairs + 3) / 4;             // 2 col groups -> 4 RG/blk
  const int grid32   = (rowPairs + 7) / 8;             // 1 col group  -> 8 RG/blk

  // adj -> bf16 once (amortized over 3 GEMM passes)
  conv_adj_bf16<<<8192, 256, 0, stream>>>(adj, adjB);

  // layer 1: h = relu(adj @ (x[:, :64] @ gw1) + gb1)
  make_b_frag<<<(KPAD * GHID + 255) / 256, 256, 0, stream>>>(x, FT, N_ROWS, DT, gw1, GHID, zfrag);
  gemm_wmma_relu<128, 4><<<grid128, 256, 0, stream>>>(adjB, zfrag, gb1, h, N_ROWS);

  // layer 2: h = relu(adj @ (h @ gw2) + gb2)
  make_b_frag<<<(KPAD * GHID + 255) / 256, 256, 0, stream>>>(h, GHID, N_ROWS, GHID, gw2, GHID, zfrag);
  gemm_wmma_relu<128, 4><<<grid128, 256, 0, stream>>>(adjB, zfrag, gb2, h, N_ROWS);

  // layer 3: h3 = relu(adj @ (h @ gw3) + gb3)
  make_b_frag<<<(KPAD * GC + 255) / 256, 256, 0, stream>>>(h, GHID, N_ROWS, GHID, gw3, GC, zfrag);
  gemm_wmma_relu<32, 2><<<grid32, 256, 0, stream>>>(adjB, zfrag, gb3, h3, N_ROWS);

  // MLP head + thresholding
  mlp_head<<<(N_ROWS + 255) / 256, 256, 0, stream>>>(h3, x, mw1, mb1, mw2, mb2, mw3, mb3, s, N_ROWS);
  reduce_min_k<<<40, 256, 0, stream>>>(s, N_ROWS, partial);
  reduce_min_k<<<1, 256, 0, stream>>>(partial, 40, minv);
  sprime_k<<<(N_ROWS + 255) / 256, 256, 0, stream>>>(s, x, minv, N_ROWS, sp);
  rank_thr_k<<<(N_ROWS + 255) / 256, 256, 0, stream>>>(sp, N_ROWS, thr);
  final_out_k<<<(N_ROWS + 255) / 256, 256, 0, stream>>>(sp, thr, N_ROWS, out);
}
